// FeatureExtractor_7438883357035
// MI455X (gfx1250) — matrix operand order repaired and checked by reading, compiled, not run on hardware
//
#include <hip/hip_runtime.h>
#include <math.h>

// ---------------------------------------------------------------------------
// CDNA5 WMMA (wave32) types: 16x16x32 bf16 -> f32 accumulate
// ---------------------------------------------------------------------------
typedef __attribute__((ext_vector_type(16))) __bf16 v16bf;
typedef __attribute__((ext_vector_type(8)))  float  v8f;

typedef unsigned int uint32x4 __attribute__((ext_vector_type(4)));
typedef int          int32x8  __attribute__((ext_vector_type(8)));
typedef int          int32x4  __attribute__((ext_vector_type(4)));

#if defined(__HIP_DEVICE_COMPILE__) && defined(__has_builtin)
#if __has_builtin(__builtin_amdgcn_tensor_load_to_lds)
#define HAVE_TDM 1
#else
#define HAVE_TDM 0
#endif
#else
#define HAVE_TDM 0
#endif

// Native bf16 convert: lets the backend emit the hardware cvt instead of a
// 4-op integer RNE sequence.
static __device__ __forceinline__ __bf16 f2bf(float f) { return (__bf16)f; }

#if HAVE_TDM
// Tensor Data Mover: 2D tile load (32 k x 64 rows of f32) into LDS,
// with LDS padding of 1 DWORD every 32 DWORDs -> lands in Asf[64][33].
// D# layout per CDNA5 ISA ch. 8 (group0: count/lds/global/type, group1: dims).
static __device__ __forceinline__ void tdm_load_a(const float* gsrc, unsigned lds_off,
                                                  int remK, int remRows, int rowStride)
{
    unsigned long long ga = (unsigned long long)(const void*)gsrc;
    uint32x4 g0;
    g0[0] = 1u;                                          // count=1
    g0[1] = lds_off;                                     // lds_addr (bytes)
    g0[2] = (unsigned)(ga & 0xffffffffull);              // global_addr[31:0]
    g0[3] = (unsigned)((ga >> 32) & 0x1ffffffull) | (2u << 30);  // addr[56:32], type=2
    unsigned td0 = (unsigned)remK;                       // tensor_dim0 (elems from tile start)
    unsigned td1 = (unsigned)remRows;                    // tensor_dim1
    unsigned long long st0 = (unsigned long long)(unsigned)rowStride;
    int32x8 g1;
    g1[0] = (int)((2u << 16) | (1u << 20) | (4u << 22)); // data_size=4B, pad_en, interval=32dw, amount=1dw
    g1[1] = (int)((td0 & 0xffffu) << 16);                // tensor_dim0[15:0]
    g1[2] = (int)(((td0 >> 16) & 0xffffu) | ((td1 & 0xffffu) << 16));
    g1[3] = (int)(((td1 >> 16) & 0xffffu) | (32u << 16));// tile_dim0 = 32
    g1[4] = (int)(64u);                                  // tile_dim1 = 64, tile_dim2 = 0
    g1[5] = (int)(st0 & 0xffffffffull);                  // tensor_dim0_stride[31:0]
    g1[6] = (int)((st0 >> 32) & 0xffffull);              // stride[47:32]
    g1[7] = 0;
    int32x4 z4 = {0, 0, 0, 0};
#if __clang_major__ >= 23
    int32x8 z8 = {0, 0, 0, 0, 0, 0, 0, 0};
    __builtin_amdgcn_tensor_load_to_lds(g0, g1, z4, z4, z8, 0);
#else
    __builtin_amdgcn_tensor_load_to_lds(g0, g1, z4, z4, 0);
#endif
}
#endif

// ---------------------------------------------------------------------------
// Implicit-GEMM conv + fused BN (+ residual) (+ ReLU).
// Block = 128 threads (4 waves), tile = 64 cout x 64 pixels, K-step = 32.
// Wave w owns cout rows [w*16, w*16+16); reuses its A fragment across
// 4 B fragments -> 4 x v_wmma_f32_16x16x32_bf16 per K-step.
// A (weights) arrives via TDM (tensor_load_to_lds) when available.
// Templated on (KH*KW, KW) so im2col index math uses constant divisors.
// All Cout in this network are multiples of 64 -> no co guards needed.
// ---------------------------------------------------------------------------
template <int KHW_T, int KW_T>
__global__ __launch_bounds__(128)
void conv_bn_wmma(const float* __restrict__ in, const float* __restrict__ wgt,
                  float* __restrict__ out, const float* __restrict__ residual,
                  const float* __restrict__ bn_g, const float* __restrict__ bn_b,
                  const float* __restrict__ bn_m, const float* __restrict__ bn_v,
                  int Cin, int H, int Wd, int Cout, int Ho, int Wo,
                  int stride, int pad, int do_relu)
{
    __shared__ float  Asf[64][33];     // [co][k] f32, 1-DWORD row pad (TDM pad feature)
    __shared__ __bf16 BsT[64][40];     // [pixel][k] bf16, rows 80B (16B-aligned chunks)

    const int tid  = threadIdx.x;
    const int lane = tid & 31;
    const int wid  = tid >> 5;
    const int half = lane >> 4;
    const int mrow = lane & 15;

    const int coBase = blockIdx.x * 64;
    const int sBase  = blockIdx.y * 64;
    const int b      = blockIdx.z;
    const int HWo    = Ho * Wo;
    const int Ktot   = Cin * KHW_T;

    // B-staging mapping: thread owns one pixel and 16 consecutive k rows.
    const int npix = tid & 63;
    const int kgrp = tid >> 6;                   // 0..1
    const int s    = sBase + npix;
    const int sOK  = (s < HWo) ? 1 : 0;
    const int sc   = sOK ? s : 0;
    const int oh   = sc / Wo;
    const int ow   = sc - oh * Wo;
    const int ihb  = oh * stride - pad;
    const int iwb  = ow * stride - pad;
    const size_t inBase = (size_t)b * Cin;

    v8f acc[4] = {{}, {}, {}, {}};

    for (int k0 = 0; k0 < Ktot; k0 += 32) {
#if HAVE_TDM
        if (tid == 0)
            tdm_load_a(wgt + (size_t)coBase * Ktot + k0,
                       (unsigned)(size_t)&Asf[0][0],
                       Ktot - k0, Cout - coBase, Ktot);
#else
        {   // manual A staging: thread t -> row t/2, 16 consecutive k (branch-free)
            int row   = tid >> 1;
            int kh16  = (tid & 1) * 16;
            const float* wrow = wgt + (size_t)(coBase + row) * Ktot;
            #pragma unroll
            for (int j = 0; j < 16; ++j) {
                int kk  = k0 + kh16 + j;
                int kv  = (kk < Ktot) ? 1 : 0;
                float a = wrow[kv ? kk : (Ktot - 1)];
                Asf[row][kh16 + j] = kv ? a : 0.0f;
            }
        }
#endif
        // ---- B staging: im2col gather, branch-free, constant-divisor math ----
        #pragma unroll
        for (int j = 0; j < 16; ++j) {
            int kb  = kgrp * 16 + j;
            int kk  = k0 + kb;
            int kv  = (kk < Ktot) ? 1 : 0;
            int kkc = kv ? kk : 0;
            int ci  = kkc / KHW_T;               // constant divisor -> mul/shift
            int rr  = kkc - ci * KHW_T;
            int kh  = rr / KW_T;
            int kw2 = rr - kh * KW_T;
            int ih  = ihb + kh;
            int iw  = iwb + kw2;
            int ok  = kv & sOK & (ih >= 0) & (ih < H) & (iw >= 0) & (iw < Wd);
            int ihc = ok ? ih : 0;
            int iwc = ok ? iw : 0;
            int cic = ok ? ci : 0;
            float v = in[((inBase + cic) * H + ihc) * Wd + iwc];
            BsT[npix][kb] = f2bf(ok ? v : 0.0f);
        }
#if HAVE_TDM
        if (tid < 32) __builtin_amdgcn_s_wait_tensorcnt(0);   // wave 0 fences its TDM
#endif
        __syncthreads();

        // ---- fragments in contiguous k order (k = 16*half + s), same for A and B ----
        v16bf afrag;
        const float* arow = &Asf[wid * 16 + mrow][0];
        #pragma unroll
        for (int r = 0; r < 8; ++r) {
            int kb = 2 * r + half * 16;
            afrag[2 * r]     = f2bf(arow[kb]);
            afrag[2 * r + 1] = f2bf(arow[kb + 1]);
        }
        #pragma unroll
        for (int q = 0; q < 4; ++q) {
            v16bf bfrag = *(const v16bf*)&BsT[q * 16 + mrow][half * 16]; // 32B contiguous
            acc[q] = __builtin_amdgcn_wmma_f32_16x16x32_bf16(
                false, afrag, false, bfrag, (short)0, acc[q], false, false);
        }
        __syncthreads();
    }

    // ---- fused epilogue: BN (+ residual) (+ ReLU), scale/shift hoisted ----
    const int coW = coBase + wid * 16;
    #pragma unroll
    for (int r = 0; r < 8; ++r) {
        int co = coW + half * 8 + r;
        float scale = bn_g[co] * rsqrtf(bn_v[co] + 1e-5f);
        float shift = bn_b[co] - bn_m[co] * scale;
        size_t rowOff = ((size_t)b * Cout + co) * HWo;
        #pragma unroll
        for (int q = 0; q < 4; ++q) {
            int ss = sBase + q * 16 + mrow;
            if (ss < HWo) {
                float y = acc[q][r] * scale + shift;
                if (residual) y += residual[rowOff + ss];
                if (do_relu)  y = fmaxf(y, 0.0f);
                out[rowOff + ss] = y;
            }
        }
    }
}

// ---------------------------------------------------------------------------
// Soft Gaussian histogram -> pdf -> cdf. One block per (b,c); one bin/thread.
// ---------------------------------------------------------------------------
__global__ __launch_bounds__(256)
void soft_cdf_kernel(const float* __restrict__ x, float* __restrict__ out,
                     int HW, int C, int rowlen)
{
    __shared__ float xs[256];
    __shared__ float sh[256];
    const int b   = blockIdx.x / C;
    const int c   = blockIdx.x % C;
    const int tid = threadIdx.x;

    const float center = (float)tid * (1.0f / 255.0f);   // linspace(0,1,256)
    const float inv_s2 = 1.0f / (0.01f * 0.01f);
    const float* px    = x + ((size_t)b * C + c) * HW;

    float hist = 0.0f;
    for (int base = 0; base < HW; base += 256) {
        int i = base + tid;
        xs[tid] = (i < HW) ? px[i] : -1.0e9f;   // sentinel -> exp == 0
        __syncthreads();
        int lim = HW - base; if (lim > 256) lim = 256;
        for (int j = 0; j < lim; ++j) {
            float d = xs[j] - center;
            hist += __expf(-d * d * inv_s2);
        }
        __syncthreads();
    }

    xs[tid] = hist;
    __syncthreads();
    for (int o = 128; o > 0; o >>= 1) {
        if (tid < o) xs[tid] += xs[tid + o];
        __syncthreads();
    }
    float tot = xs[0];
    __syncthreads();

    sh[tid] = hist / (tot + 1e-6f);
    __syncthreads();
    for (int o = 1; o < 256; o <<= 1) {
        float t = (tid >= o) ? sh[tid - o] : 0.0f;
        __syncthreads();
        sh[tid] += t;
        __syncthreads();
    }
    out[(size_t)b * rowlen + (size_t)c * 256 + tid] = sh[tid];
}

// ---------------------------------------------------------------------------
// 3x3 stride-2 pad-1 max pool, one thread per output element.
// ---------------------------------------------------------------------------
__global__ void maxpool_kernel(const float* __restrict__ in, float* __restrict__ out,
                               int C, int H, int W, int Ho, int Wo, int total)
{
    int idx = blockIdx.x * blockDim.x + threadIdx.x;
    if (idx >= total) return;
    int ow = idx % Wo; int t = idx / Wo;
    int oh = t % Ho;   t /= Ho;
    int c  = t % C;    int b = t / C;
    float m = -3.4e38f;
    for (int kh = 0; kh < 3; ++kh)
        for (int kw = 0; kw < 3; ++kw) {
            int ih = oh * 2 - 1 + kh, iw = ow * 2 - 1 + kw;
            if (ih >= 0 && ih < H && iw >= 0 && iw < W)
                m = fmaxf(m, in[(((size_t)b * C + c) * H + ih) * W + iw]);
        }
    out[idx] = m;
}

// ---------------------------------------------------------------------------
// Global average pool (2,512,7,7) -> out[b*rowlen + off + c]
// ---------------------------------------------------------------------------
__global__ __launch_bounds__(64)
void gap_kernel(const float* __restrict__ in, float* __restrict__ out,
                int HW, int Cc, int rowlen, int off)
{
    __shared__ float sh[64];
    int b = blockIdx.x / Cc, c = blockIdx.x % Cc;
    float s = 0.0f;
    for (int i = threadIdx.x; i < HW; i += 64)
        s += in[((size_t)b * Cc + c) * HW + i];
    sh[threadIdx.x] = s;
    __syncthreads();
    for (int o = 32; o > 0; o >>= 1) {
        if (threadIdx.x < o) sh[threadIdx.x] += sh[threadIdx.x + o];
        __syncthreads();
    }
    if (threadIdx.x == 0)
        out[(size_t)b * rowlen + off + c] = sh[0] / (float)HW;
}

// ---------------------------------------------------------------------------
// Host-side orchestration
// ---------------------------------------------------------------------------
struct BNP { const float *g, *b, *m, *v; };

extern "C" void kernel_launch(void* const* d_in, const int* in_sizes, int n_in,
                              void* d_out, int out_size, void* d_ws, size_t ws_size,
                              hipStream_t stream)
{
    (void)in_sizes; (void)n_in; (void)out_size; (void)ws_size;
    const int B = 2, Cimg = 3;
    const int ROWLEN = 3 * 256 + 512;   // 1280

    // ---- walk flattened params (jax tree order: sorted dict keys) ----
    int cur = 0;
    auto nf = [&]() { return (const float*)d_in[cur++]; };
    const float* x = nf();
    BNP bn_stem; bn_stem.b = nf(); bn_stem.g = nf(); bn_stem.m = nf(); bn_stem.v = nf();
    const float* w_stem = nf();

    struct Blk { BNP bn1, bn2, dbn; const float *c1, *c2, *down; bool has_down; };
    Blk blks[4][2];
    const int couts[4]   = {64, 128, 256, 512};
    const int strides[4] = {1, 2, 2, 2};
    int cin = 64;
    for (int si = 0; si < 4; ++si)
        for (int bi = 0; bi < 2; ++bi) {
            Blk& k = blks[si][bi];
            int s = (bi == 0) ? strides[si] : 1;
            k.has_down = (s != 1 || cin != couts[si]);
            k.bn1.b = nf(); k.bn1.g = nf(); k.bn1.m = nf(); k.bn1.v = nf();
            k.bn2.b = nf(); k.bn2.g = nf(); k.bn2.m = nf(); k.bn2.v = nf();
            k.c1 = nf(); k.c2 = nf();
            if (k.has_down) {
                k.dbn.b = nf(); k.dbn.g = nf(); k.dbn.m = nf(); k.dbn.v = nf();
                k.down = nf();
            } else {
                k.down = nullptr;
            }
            cin = couts[si];
        }

    // ---- workspace carve-out (whole working set is L2-resident on MI455X) ----
    float* WS = (float*)d_ws;
    size_t off = 0;
    auto alloc = [&](size_t n) { float* p = WS + off; off += n; return p; };
    float* stem = alloc((size_t)B * 64 * 112 * 112);
    float* bufA = alloc((size_t)B * 64 * 56 * 56);
    float* bufB = alloc((size_t)B * 64 * 56 * 56);
    float* bufT = alloc((size_t)B * 64 * 56 * 56);
    float* bufS = alloc((size_t)B * 64 * 56 * 56);

    auto conv = [&](const float* inp, int Cin_, int H_, const float* w_, BNP bn,
                    int Cout_, int K_, int s_, int p_, float* outp,
                    const float* res, int relu) {
        int Ho = (H_ + 2 * p_ - K_) / s_ + 1;
        dim3 grid((Cout_ + 63) / 64, (Ho * Ho + 63) / 64, B);
        if (K_ == 7)
            conv_bn_wmma<49, 7><<<grid, 128, 0, stream>>>(inp, w_, outp, res,
                bn.g, bn.b, bn.m, bn.v, Cin_, H_, H_, Cout_, Ho, Ho, s_, p_, relu);
        else if (K_ == 3)
            conv_bn_wmma<9, 3><<<grid, 128, 0, stream>>>(inp, w_, outp, res,
                bn.g, bn.b, bn.m, bn.v, Cin_, H_, H_, Cout_, Ho, Ho, s_, p_, relu);
        else
            conv_bn_wmma<1, 1><<<grid, 128, 0, stream>>>(inp, w_, outp, res,
                bn.g, bn.b, bn.m, bn.v, Cin_, H_, H_, Cout_, Ho, Ho, s_, p_, relu);
        return Ho;
    };

    // ---- branch 1: differentiable CDF ----
    soft_cdf_kernel<<<B * Cimg, 256, 0, stream>>>(x, (float*)d_out, 224 * 224, Cimg, ROWLEN);

    // ---- branch 2: ResNet-18 ----
    conv(x, 3, 224, w_stem, bn_stem, 64, 7, 2, 3, stem, nullptr, 1);   // -> 112x112
    int total = B * 64 * 56 * 56;
    maxpool_kernel<<<(total + 255) / 256, 256, 0, stream>>>(stem, bufA, 64, 112, 112, 56, 56, total);

    float* curb = bufA; float* nxt = bufB;
    int cinr = 64, Hc = 56;
    for (int si = 0; si < 4; ++si)
        for (int bi = 0; bi < 2; ++bi) {
            Blk& k = blks[si][bi];
            int cout = couts[si];
            int s = (bi == 0) ? strides[si] : 1;
            int Ho = conv(curb, cinr, Hc, k.c1, k.bn1, cout, 3, s, 1, bufT, nullptr, 1);
            const float* res;
            if (k.has_down) {
                conv(curb, cinr, Hc, k.down, k.dbn, cout, 1, s, 0, bufS, nullptr, 0);
                res = bufS;
            } else {
                res = curb;
            }
            conv(bufT, cout, Ho, k.c2, k.bn2, cout, 3, 1, 1, nxt, res, 1);
            float* t = curb; curb = nxt; nxt = t;
            cinr = cout; Hc = Ho;
        }

    gap_kernel<<<B * 512, 64, 0, stream>>>(curb, (float*)d_out, 7 * 7, 512, ROWLEN, 768);
}
